// TinyScaleDotProductAttention_4844723109980
// MI455X (gfx1250) — compile-verified, matrix operand order checked by reading, not run on hardware
//
#include <hip/hip_runtime.h>

typedef __attribute__((ext_vector_type(16))) __bf16 v16bf;
typedef __attribute__((ext_vector_type(8)))  __bf16 v8bf;
typedef __attribute__((ext_vector_type(4)))  __bf16 v4bf;
typedef __attribute__((ext_vector_type(8)))  float  v8f;

#define SEQ   2048
#define DHEAD 128
#define BM    128    // query rows per block
#define BN    64     // key cols per tile
#define NWAVE 8
#define NT    256

__global__ __launch_bounds__(NT)
void fa_fwd_causal(const float* __restrict__ Q, const float* __restrict__ K,
                   const float* __restrict__ V, float* __restrict__ O) {
  __shared__ __bf16 sK [BN * DHEAD];      // [key][d] row-major
  __shared__ __bf16 sVt[DHEAD * BN];      // [d][key] transposed
  __shared__ __bf16 sP [NWAVE][16 * BN];  // per-wave P staging [m][n]

  const int tid  = threadIdx.x;
  const int wave = tid >> 5;
  const int lane = tid & 31;
  const int l16  = lane & 15;
  const int hi   = lane >> 4;             // 0: lanes 0-15, 1: lanes 16-31
  const int dk   = hi ? 8 : 0;            // k-offset for A/B fragment halves

  const int mtiles = SEQ / BM;            // 16
  const int bh     = blockIdx.x / mtiles;
  const int m0     = (blockIdx.x % mtiles) * BM;
  const int qrow0  = m0 + wave * 16;

  const size_t base = (size_t)bh * SEQ * DHEAD;
  const float* gq = Q + base;
  const float* gk = K + base;
  const float* gv = V + base;
  float*       go = O + base;

  const float scale = 0.08838834764831845f; // 1/sqrt(128)

  // ---- preload Q fragments (A-matrix layout), scale folded into bf16 convert
  v16bf qf[4];
  {
    const float* qrow = gq + (size_t)(qrow0 + l16) * DHEAD;
    for (int c = 0; c < 4; ++c) {
      union { v16bf v; __bf16 e[16]; } u;
      for (int h = 0; h < 2; ++h) {
        const float4* p = (const float4*)(qrow + 32 * c + dk + 16 * h);
        float4 a = p[0], b = p[1];
        u.e[8*h+0] = (__bf16)(a.x * scale); u.e[8*h+1] = (__bf16)(a.y * scale);
        u.e[8*h+2] = (__bf16)(a.z * scale); u.e[8*h+3] = (__bf16)(a.w * scale);
        u.e[8*h+4] = (__bf16)(b.x * scale); u.e[8*h+5] = (__bf16)(b.y * scale);
        u.e[8*h+6] = (__bf16)(b.z * scale); u.e[8*h+7] = (__bf16)(b.w * scale);
      }
      qf[c] = u.v;
    }
  }

  // all-ones B fragment: P x ones -> row sums in accumulator tile 8
  v16bf ones;
  {
    union { v16bf v; __bf16 e[16]; } u;
    for (int i = 0; i < 16; ++i) u.e[i] = (__bf16)1.0f;
    ones = u.v;
  }

  v8f acc[9];                              // 8 d-tiles + 1 rowsum tile
  for (int j = 0; j < 9; ++j)
    for (int i = 0; i < 8; ++i) acc[j][i] = 0.f;
  float mrow[8];
  for (int i = 0; i < 8; ++i) mrow[i] = -1e30f;

  const int ntiles = (m0 + BM) / BN;      // causal bound for this block

  for (int kt = 0; kt < ntiles; ++kt) {
    const int k0 = kt * BN;
    __syncthreads();
    // ---- cooperative stage: K row-major bf16 (vectorized), V transposed bf16
    for (int i = 0; i < 8; ++i) {
      int g   = tid + i * NT;             // [0, 2048)
      int row = g >> 5;                   // K: 32 float4 per 128-wide row
      int c4  = (g & 31) << 2;
      float4 kq = ((const float4*)(gk + (size_t)(k0 + row) * DHEAD))[g & 31];
      union { v4bf v; __bf16 e[4]; } kb;
      kb.e[0] = (__bf16)kq.x; kb.e[1] = (__bf16)kq.y;
      kb.e[2] = (__bf16)kq.z; kb.e[3] = (__bf16)kq.w;
      *(v4bf*)&sK[row * DHEAD + c4] = kb.v;
      // V: coalesced dword loads along d, packed b64 transposed stores
      int d  = g & 127;
      int k4 = (g >> 7) << 2;             // key group of 4 (0..60)
      union { v4bf v; __bf16 e[4]; } vb;
      for (int j = 0; j < 4; ++j)
        vb.e[j] = (__bf16)gv[(size_t)(k0 + k4 + j) * DHEAD + d];
      *(v4bf*)&sVt[d * BN + k4] = vb.v;
    }
    // prefetch next K/V tile into cache while this tile is consumed
    if (kt + 1 < ntiles) {
      int row = tid >> 2;                 // 64 rows x 4 x 128B
      int col = (tid & 3) << 5;
      __builtin_prefetch(gk + (size_t)(k0 + BN + row) * DHEAD + col, 0, 1);
      __builtin_prefetch(gv + (size_t)(k0 + BN + row) * DHEAD + col, 0, 1);
    }
    __syncthreads();

    // per-wave causal skip: tile entirely above this wave's rows contributes 0
    if (k0 > qrow0 + 15) continue;
    const bool diag = (k0 + BN - 1 > qrow0); // masking only on diagonal tile

    // ---- scores: S = (Q*scale) * K^T, four 16x16 tiles, 16 WMMAs
    v8f s[4];
    for (int t = 0; t < 4; ++t) {
      for (int i = 0; i < 8; ++i) s[t][i] = 0.f;
      const __bf16* kb = &sK[(16 * t + l16) * DHEAD + dk];
      for (int c = 0; c < 4; ++c) {
        union { v16bf v; v8bf h[2]; } b;
        b.h[0] = *(const v8bf*)(kb + 32 * c);
        b.h[1] = *(const v8bf*)(kb + 32 * c + 16);
        s[t] = __builtin_amdgcn_wmma_f32_16x16x32_bf16(
                   false, qf[c], false, b.v, (short)0, s[t], false, false);
      }
    }

    // ---- causal mask (diag tiles only) + online softmax (max via shuffles)
    float p[4][8];
    for (int i = 0; i < 8; ++i) {
      float a[4];
      for (int t = 0; t < 4; ++t) a[t] = s[t][i];
      if (diag) {
        int r = qrow0 + i + 8 * hi;
        for (int t = 0; t < 4; ++t)
          a[t] = (k0 + 16 * t + l16 > r) ? -1e30f : a[t];
      }
      float v = fmaxf(fmaxf(a[0], a[1]), fmaxf(a[2], a[3]));
      for (int off = 1; off < 16; off <<= 1)
        v = fmaxf(v, __shfl_xor(v, off, 32));
      float mn    = fmaxf(mrow[i], v);
      float alpha = __expf(mrow[i] - mn);
      mrow[i] = mn;
      for (int t = 0; t < 4; ++t) p[t][i] = __expf(a[t] - mn);
      for (int j = 0; j < 9; ++j) acc[j][i] *= alpha;
    }

    // ---- C-layout f32 P -> A-layout bf16 via per-wave LDS bounce
    __bf16* pw = sP[wave];
    for (int i = 0; i < 8; ++i) {
      int r = i + 8 * hi;
      for (int t = 0; t < 4; ++t)
        pw[r * BN + 16 * t + l16] = (__bf16)p[t][i];
    }
    asm volatile("s_wait_dscnt 0" ::: "memory");  // wave-local LDS RAW
    union { v16bf v; v8bf h[2]; } pa0, pa1;       // P A-frags, k=0..31 / 32..63
    pa0.h[0] = *(const v8bf*)&pw[l16 * BN + dk];
    pa0.h[1] = *(const v8bf*)&pw[l16 * BN + dk + 16];
    pa1.h[0] = *(const v8bf*)&pw[l16 * BN + dk + 32];
    pa1.h[1] = *(const v8bf*)&pw[l16 * BN + dk + 48];

    // ---- acc += P * V, 16 WMMAs over d-tiles; +2 WMMAs for row sums
    for (int j = 0; j < 8; ++j) {
      const __bf16* vrow = &sVt[(16 * j + l16) * BN];
      union { v16bf v; v8bf h[2]; } b0, b1;
      b0.h[0] = *(const v8bf*)(vrow + dk);
      b0.h[1] = *(const v8bf*)(vrow + dk + 16);
      b1.h[0] = *(const v8bf*)(vrow + dk + 32);
      b1.h[1] = *(const v8bf*)(vrow + dk + 48);
      acc[j] = __builtin_amdgcn_wmma_f32_16x16x32_bf16(
                   false, pa0.v, false, b0.v, (short)0, acc[j], false, false);
      acc[j] = __builtin_amdgcn_wmma_f32_16x16x32_bf16(
                   false, pa1.v, false, b1.v, (short)0, acc[j], false, false);
    }
    acc[8] = __builtin_amdgcn_wmma_f32_16x16x32_bf16(
                 false, pa0.v, false, ones, (short)0, acc[8], false, false);
    acc[8] = __builtin_amdgcn_wmma_f32_16x16x32_bf16(
                 false, pa1.v, false, ones, (short)0, acc[8], false, false);
  }

  // ---- epilogue: normalize by rowsum tile and store f32
  for (int i = 0; i < 8; ++i) {
    int r = qrow0 + i + 8 * hi;
    float inv = 1.f / acc[8][i];
    for (int j = 0; j < 8; ++j)
      go[(size_t)r * DHEAD + 16 * j + l16] = acc[j][i] * inv;
  }
}

extern "C" void kernel_launch(void* const* d_in, const int* in_sizes, int n_in,
                              void* d_out, int out_size, void* d_ws, size_t ws_size,
                              hipStream_t stream) {
  (void)in_sizes; (void)n_in; (void)out_size; (void)d_ws; (void)ws_size;
  const float* q = (const float*)d_in[0];
  const float* k = (const float*)d_in[1];
  const float* v = (const float*)d_in[2];
  float* o = (float*)d_out;
  const int B = 2, H = 16;
  dim3 grid(B * H * (SEQ / BM));
  dim3 block(NT);
  hipLaunchKernelGGL(fa_fwd_causal, grid, block, 0, stream, q, k, v, o);
}
